// Mutual_Information_23708219474514
// MI455X (gfx1250) — matrix operand order repaired and checked by reading, not compiled, not run on hardware
//
#include <hip/hip_runtime.h>
#include <hip/hip_fp16.h>

typedef __attribute__((ext_vector_type(16))) _Float16 v16h;
typedef __attribute__((ext_vector_type(2)))  _Float16 h2;
typedef __attribute__((ext_vector_type(8)))  float    v8f;

#define BINS      50
#define PAD_BINS  64
#define SIGMA_C   10.0f
#define DELTA_C   (1.0f / 50.0f)          // (MAX_V - MIN_V) / BINS
#define EPS_C     1e-8f
#define D_TOT     (384 * 384)             // 147456 pixels per image
#define NBATCH    8
#define SPLITK    72
#define SLICE     (D_TOT / SPLITK)        // 2048
#define KC        256                     // pixels per chunk (8 WMMA k-steps)
#define NSTEP     (KC / 32)               // 8
#define NCHUNK    (SLICE / KC)            // 8

__device__ __forceinline__ float fast_sigmoid(float z) {
    // sigmoid(z) = 1 / (1 + exp(-z)) ; v_exp_f32 + v_rcp_f32
    float e = __expf(-z);
    return __builtin_amdgcn_rcpf(1.0f + e);
}

// ---------------------------------------------------------------------------
// Kernel 1: fused soft-bin + split-K WMMA Gram matrix.
//
// soft_bin(v, b) = sigmoid(S*(v - b*D)) - sigmoid(S*(v - (b+1)*D)):
// edges telescope, so one thread walking all 50 bins of one pixel needs only
// 51 sigmoids (vs 100 naive). Threads own PIXEL PAIRS: adjacent pixels are the
// lo/hi halves of the same fragment dword, so each bin is one ds_store_b32.
//
// grid = (8 batches, SPLITK slabs), block = 256 (8 waves).
// Waves 0-3 generate A (im1 soft-bins), waves 4-7 generate B (im2 soft-bins);
// every wave then consumes fragments: wave w owns output tiles
// (row-block rb = w&3) x (col tiles 2*(w>>2), 2*(w>>2)+1).
// ---------------------------------------------------------------------------
__global__ void __launch_bounds__(256) hgram_kernel(const float* __restrict__ im1,
                                                    const float* __restrict__ im2,
                                                    float* __restrict__ hg) {
    // Fragment layout: [kstep(8)][tileblk(4)][lane(32)][8 dwords] per matrix
    __shared__ h2 shA[NSTEP * 4 * 32 * 8] __attribute__((aligned(32)));  // 32 KB
    __shared__ h2 shB[NSTEP * 4 * 32 * 8] __attribute__((aligned(32)));  // 32 KB

    const int tid   = threadIdx.x;
    const int lane  = tid & 31;
    const int wave  = tid >> 5;
    const int n     = blockIdx.x;
    const int slab  = blockIdx.y;
    const int rb    = wave & 3;       // A row-block (bins rb*16 .. rb*16+15)
    const int cpair = wave >> 2;      // B col-pair (tiles 2*cpair, 2*cpair+1)

    // ---- producer role (wave-uniform): waves 0-3 -> A, waves 4-7 -> B ----
    const int isB  = tid >> 7;        // 0 = A (im1), 1 = B (im2)
    const int pairIdx = tid & 127;    // pixel pair 0..127 within chunk
    const int k0   = pairIdx * 2;     // first pixel of pair
    const int s    = k0 >> 5;         // WMMA k-step 0..7
    const int kk   = k0 & 31;         // k within 16x16x32 tile (even)

    // dword slot for (kk, kk+1) — identical for both halves of the pair.
    // A-matrix (16-bit 16x32) layout per ISA 7.12.2:
    const int alane = ((kk >> 3) & 1) << 4;
    const int adw   = ((kk >> 4) << 2) + ((kk & 7) >> 1);
    // B-matrix (32x16, same K order as A):
    const int blane = ((kk >> 3) & 1) << 4;
    const int bdw   = ((kk >> 4) << 2) + ((kk & 7) >> 1);

    h2* myBuf        = isB ? shB : shA;
    const int mylane = isB ? blane : alane;
    const int mydw   = isB ? bdw : adw;

    // Pre-zero pad bins 50..63 (their slots are never rewritten in the loop).
#pragma unroll
    for (int b = BINS; b < PAD_BINS; ++b) {
        myBuf[((s * 4 + (b >> 4)) * 32 + ((b & 15) + mylane)) * 8 + mydw] = (h2)0.0f;
    }

    const float* __restrict__ src =
        (isB ? im2 : im1) + n * D_TOT + slab * SLICE;

    v8f acc0 = {};
    v8f acc1 = {};

    for (int chunk = 0; chunk < NCHUNK; ++chunk) {
        const float2 v2 = *(const float2*)&src[chunk * KC + k0];

        // Telescoping edges: e_b = sigmoid(S*(v - b*D)), bin value = e_b - e_{b+1}
        float e0 = fast_sigmoid(SIGMA_C * v2.x);
        float e1 = fast_sigmoid(SIGMA_C * v2.y);
#pragma unroll
        for (int b = 0; b < BINS; ++b) {
            const float edge = (float)(b + 1) * DELTA_C;
            const float f0 = fast_sigmoid(SIGMA_C * (v2.x - edge));
            const float f1 = fast_sigmoid(SIGMA_C * (v2.y - edge));
            h2 packed;
            packed.x = (_Float16)(e0 - f0);
            packed.y = (_Float16)(e1 - f1);
            myBuf[((s * 4 + (b >> 4)) * 32 + ((b & 15) + mylane)) * 8 + mydw] = packed;
            e0 = f0;
            e1 = f1;
        }
        __syncthreads();

#pragma unroll
        for (int s2 = 0; s2 < NSTEP; ++s2) {
            const v16h a  = *(const v16h*)&shA[(((s2 * 4 + rb) * 32) + lane) * 8];
            const v16h b0 = *(const v16h*)&shB[(((s2 * 4 + cpair * 2)     * 32) + lane) * 8];
            const v16h b1 = *(const v16h*)&shB[(((s2 * 4 + cpair * 2 + 1) * 32) + lane) * 8];
            acc0 = __builtin_amdgcn_wmma_f32_16x16x32_f16(false, a, false, b0,
                                                          (short)0, acc0, false, false);
            acc1 = __builtin_amdgcn_wmma_f32_16x16x32_f16(false, a, false, b1,
                                                          (short)0, acc1, false, false);
        }
        __syncthreads();
    }

    // C/D layout: VGPR r -> (M = r + 8*(lane>=16), N = lane&15)
    const int M0  = (lane >> 4) << 3;
    const int col = lane & 15;
    float* base = hg + n * PAD_BINS * PAD_BINS;
#pragma unroll
    for (int r = 0; r < 8; ++r) {
        const int row = rb * 16 + r + M0;
        unsafeAtomicAdd(&base[row * PAD_BINS + (cpair * 2)     * 16 + col], acc0[r]);
        unsafeAtomicAdd(&base[row * PAD_BINS + (cpair * 2 + 1) * 16 + col], acc1[r]);
    }
}

// ---------------------------------------------------------------------------
// Kernel 2: normalization + mutual-information reduction (tiny, one block).
// Padded entries (bin >= 50) are exactly zero, so including them adds 0.
// ---------------------------------------------------------------------------
__global__ void __launch_bounds__(256) mi_kernel(const float* __restrict__ hg,
                                                 float* __restrict__ out) {
    __shared__ float red[256];
    __shared__ float mx[PAD_BINS], my[PAD_BINS];
    __shared__ float s_invT;
    const int tid = threadIdx.x;

    // total over the whole batch histogram
    float t = 0.0f;
    for (int i = tid; i < NBATCH * PAD_BINS * PAD_BINS; i += 256) t += hg[i];
    red[tid] = t;
    __syncthreads();
    for (int off = 128; off > 0; off >>= 1) {
        if (tid < off) red[tid] += red[tid + off];
        __syncthreads();
    }
    if (tid == 0) s_invT = 1.0f / red[0];
    __syncthreads();
    const float invT = s_invT;

    float mi = 0.0f;
    for (int n = 0; n < NBATCH; ++n) {
        const float* base = hg + n * PAD_BINS * PAD_BINS;
        if (tid < PAD_BINS) {
            float s = 0.0f;
            for (int c = 0; c < PAD_BINS; ++c) s += base[tid * PAD_BINS + c];
            mx[tid] = s * invT;
        } else if (tid < 2 * PAD_BINS) {
            const int c = tid - PAD_BINS;
            float s = 0.0f;
            for (int r = 0; r < PAD_BINS; ++r) s += base[r * PAD_BINS + c];
            my[c] = s * invT;
        }
        __syncthreads();
        for (int e = tid; e < PAD_BINS * PAD_BINS; e += 256) {
            const int b = e >> 6, c = e & 63;
            const float p = base[e] * invT;
            const float j = mx[b] * my[c];
            mi += p * (__logf(p + EPS_C) - __logf(j + EPS_C));
        }
        __syncthreads();
    }
    red[tid] = mi;
    __syncthreads();
    for (int off = 128; off > 0; off >>= 1) {
        if (tid < off) red[tid] += red[tid + off];
        __syncthreads();
    }
    if (tid == 0) out[0] = red[0];
}

extern "C" void kernel_launch(void* const* d_in, const int* in_sizes, int n_in,
                              void* d_out, int out_size, void* d_ws, size_t ws_size,
                              hipStream_t stream) {
    const float* im1 = (const float*)d_in[0];
    const float* im2 = (const float*)d_in[1];
    float* hg = (float*)d_ws;  // [8][64][64] f32 partial-sum histogram

    hipMemsetAsync(hg, 0, NBATCH * PAD_BINS * PAD_BINS * sizeof(float), stream);
    hgram_kernel<<<dim3(NBATCH, SPLITK), 256, 0, stream>>>(im1, im2, hg);
    mi_kernel<<<1, 256, 0, stream>>>(hg, (float*)d_out);
}
